// DiTBlock_89970974916629
// MI455X (gfx1250) — compile-verified, matrix operand order checked by reading, not run on hardware
//
#include <hip/hip_runtime.h>
#include <hip/hip_bf16.h>
#include <math.h>

typedef __bf16 bf16_t;
typedef __attribute__((ext_vector_type(16))) __bf16 v16bf;
typedef __attribute__((ext_vector_type(8)))  __bf16 bf16x8;
typedef __attribute__((ext_vector_type(8)))  float  v8f;

#define H_  768
#define T_  2048
#define B_  2
#define NH_ 12
#define HD_ 64

// ---------------- fp32 -> bf16 conversion ----------------
__global__ void k_f2bf(const float* __restrict__ src, bf16_t* __restrict__ dst, int n) {
    int i = blockIdx.x * blockDim.x + threadIdx.x;
    if (i < n) dst[i] = (bf16_t)src[i];
}

// ---------------- fp32 [K,N] -> bf16 transposed [N,K] ----------------
__global__ void k_f2bf_t(const float* __restrict__ src, bf16_t* __restrict__ dst,
                         int K, int N) {
    int i = blockIdx.x * blockDim.x + threadIdx.x;
    if (i >= K * N) return;
    int k = i / N, n = i % N;
    dst[(size_t)n * K + k] = (bf16_t)src[i];
}

// ---------------- tiny fp32 GEMM for the adaLN MLP (2 rows) ----------------
// act: 0 = none, 1 = SiLU
__global__ void k_small_mm(const float* __restrict__ A, const float* __restrict__ W,
                           const float* __restrict__ bias, float* __restrict__ out,
                           int M, int K, int N, int act) {
    int idx = blockIdx.x * blockDim.x + threadIdx.x;
    if (idx >= M * N) return;
    int r = idx / N, c = idx % N;
    float acc = 0.f;
    for (int k = 0; k < K; ++k) acc = fmaf(A[r * K + k], W[(size_t)k * N + c], acc);
    acc += bias[c];
    if (act == 1) acc = acc / (1.f + expf(-acc));
    out[idx] = acc;
}

// ---------------- LayerNorm + adaLN modulate -> bf16 ----------------
__global__ void k_ln_mod(const float* __restrict__ x,
                         const float* __restrict__ g, const float* __restrict__ b,
                         const float* __restrict__ temb, int gateOff, int shiftOff,
                         bf16_t* __restrict__ out) {
    __shared__ float sbuf[256];
    int row = blockIdx.x;            // 0..B*T-1
    int batch = row / T_;
    int tid = threadIdx.x;
    const float* xr = x + (size_t)row * H_;
    const float* gate  = temb + (size_t)batch * (6 * H_) + gateOff;
    const float* shift = temb + (size_t)batch * (6 * H_) + shiftOff;

    float v0 = xr[tid], v1 = xr[tid + 256], v2 = xr[tid + 512];
    sbuf[tid] = v0 + v1 + v2;
    __syncthreads();
    for (int off = 128; off > 0; off >>= 1) {
        if (tid < off) sbuf[tid] += sbuf[tid + off];
        __syncthreads();
    }
    float mu = sbuf[0] * (1.f / H_);
    __syncthreads();
    float d0 = v0 - mu, d1 = v1 - mu, d2 = v2 - mu;
    sbuf[tid] = d0 * d0 + d1 * d1 + d2 * d2;
    __syncthreads();
    for (int off = 128; off > 0; off >>= 1) {
        if (tid < off) sbuf[tid] += sbuf[tid + off];
        __syncthreads();
    }
    float rstd = rsqrtf(sbuf[0] * (1.f / H_) + 1e-5f);

    bf16_t* orow = out + (size_t)row * H_;
    int c;
    c = tid;        orow[c] = (bf16_t)(gate[c] * (d0 * rstd * g[c] + b[c]) + shift[c]);
    c = tid + 256;  orow[c] = (bf16_t)(gate[c] * (d1 * rstd * g[c] + b[c]) + shift[c]);
    c = tid + 512;  orow[c] = (bf16_t)(gate[c] * (d2 * rstd * g[c] + b[c]) + shift[c]);
}

// ---------------- V repack: qkv[B*T,3H] (v slice) -> vT[BH, HD, T] ----------------
__global__ void k_repack_v(const bf16_t* __restrict__ qkv, bf16_t* __restrict__ vT) {
    int idx = blockIdx.x * blockDim.x + threadIdx.x;
    if (idx >= B_ * T_ * H_) return;
    int row = idx / H_, col = idx % H_;
    int h = col / HD_, d = col % HD_;
    int bb = row / T_, tt = row % T_;
    int bh = bb * NH_ + h;
    vT[((size_t)bh * HD_ + d) * T_ + tt] = qkv[(size_t)row * (3 * H_) + 2 * H_ + col];
}

// ---------------- row softmax (fp32 in, bf16 out) ----------------
__global__ void k_softmax_row(const float* __restrict__ wei, bf16_t* __restrict__ p, int n) {
    __shared__ float sbuf[256];
    int row = blockIdx.x, tid = threadIdx.x;
    const float* wr = wei + (size_t)row * n;
    float m = -1e30f;
    for (int c = tid; c < n; c += 256) m = fmaxf(m, wr[c]);
    sbuf[tid] = m; __syncthreads();
    for (int off = 128; off > 0; off >>= 1) {
        if (tid < off) sbuf[tid] = fmaxf(sbuf[tid], sbuf[tid + off]);
        __syncthreads();
    }
    m = sbuf[0]; __syncthreads();
    float s = 0.f;
    for (int c = tid; c < n; c += 256) s += expf(wr[c] - m);
    sbuf[tid] = s; __syncthreads();
    for (int off = 128; off > 0; off >>= 1) {
        if (tid < off) sbuf[tid] += sbuf[tid + off];
        __syncthreads();
    }
    float inv = 1.f / sbuf[0];
    for (int c = tid; c < n; c += 256)
        p[(size_t)row * n + c] = (bf16_t)(expf(wr[c] - m) * inv);
}

// ---------------- generic bf16 WMMA GEMM (B given transposed) ----------------
// C = epilogue(alpha * A(MxK) @ B(KxN) + bias), with BT = B^T stored row-major [N][K].
// Preconditions: M % 128 == 0, K % 32 == 0, N % 8 == 0; rows/cols 16B-aligned.
// act: 0 none, 1 exact GELU. If resid != null: out = resid + gate[row/rowsPerGate][col]*val.
// Output fp32 to Cf if non-null, else bf16 to Cb.
__global__ __launch_bounds__(256)
void k_gemm_bf16(const bf16_t* __restrict__ A, int lda,
                 const bf16_t* __restrict__ BT, int ldbt,
                 float* __restrict__ Cf, bf16_t* __restrict__ Cb, int ldc,
                 const float* __restrict__ bias, float alpha, int act,
                 const float* __restrict__ resid, const float* __restrict__ gate,
                 int gateLd, int rowsPerGate, int M, int N, int K) {
    __shared__ bf16_t As[128][40];   // 128 M-rows x 32 K (padded)
    __shared__ bf16_t Bs[128][40];   // 128 N-rows x 32 K (padded)

    int tid  = threadIdx.x;
    int lane = tid & 31, wave = tid >> 5;
    int wr = wave >> 2, wc = wave & 3;         // 2 x 4 wave grid
    int half = lane >> 4, l16 = lane & 15;
    int m0 = blockIdx.y * 128, n0 = blockIdx.x * 128;

    // Per-thread staging coordinates: 2 vectors of 8 bf16 each for A and B.
    int r0 = tid >> 2,           c0 = (tid & 3) << 3;
    int r1 = (tid + 256) >> 2,   c1 = ((tid + 256) & 3) << 3;

    v8f acc[4][2] = {};

    for (int k0 = 0; k0 < K; k0 += 32) {
        // Stage A tile (128x32): contiguous vector copies (M always % 128 == 0).
        *(bf16x8*)(&As[r0][c0]) = *(const bf16x8*)(A + (size_t)(m0 + r0) * lda + k0 + c0);
        *(bf16x8*)(&As[r1][c1]) = *(const bf16x8*)(A + (size_t)(m0 + r1) * lda + k0 + c1);
        // Stage B tile (128 N-rows x 32 K) from BT: same contiguous pattern, N-guarded.
        {
            bf16x8 d = {};
            if (n0 + r0 < N) d = *(const bf16x8*)(BT + (size_t)(n0 + r0) * ldbt + k0 + c0);
            *(bf16x8*)(&Bs[r0][c0]) = d;
            bf16x8 e = {};
            if (n0 + r1 < N) e = *(const bf16x8*)(BT + (size_t)(n0 + r1) * ldbt + k0 + c1);
            *(bf16x8*)(&Bs[r1][c1]) = e;
        }
        // Prefetch next K-tile (global_prefetch_b8 path).
        if (k0 + 32 < K) {
            __builtin_prefetch(A + (size_t)(m0 + r0) * lda + k0 + 32 + c0, 0, 3);
            if (n0 + r0 < N)
                __builtin_prefetch(BT + (size_t)(n0 + r0) * ldbt + k0 + 32 + c0, 0, 3);
        }
        __syncthreads();

        // Fragment loads per ISA 16-bit layout:
        // lane L holds 8 contiguous K elems of row (L%16); K-half chosen by L/16;
        // elements 0..7 <- K[half*8 .. +7], elements 8..15 <- K[16+half*8 .. +7].
        v16bf afr[4], bfr[2];
        #pragma unroll
        for (int i = 0; i < 4; ++i) {
            int m = wr * 64 + i * 16 + l16;
            bf16x8 lo = *(const bf16x8*)(&As[m][half * 8]);
            bf16x8 hi = *(const bf16x8*)(&As[m][16 + half * 8]);
            #pragma unroll
            for (int e = 0; e < 8; ++e) { afr[i][e] = lo[e]; afr[i][8 + e] = hi[e]; }
        }
        #pragma unroll
        for (int j = 0; j < 2; ++j) {
            int n = wc * 32 + j * 16 + l16;
            bf16x8 lo = *(const bf16x8*)(&Bs[n][half * 8]);
            bf16x8 hi = *(const bf16x8*)(&Bs[n][16 + half * 8]);
            #pragma unroll
            for (int e = 0; e < 8; ++e) { bfr[j][e] = lo[e]; bfr[j][8 + e] = hi[e]; }
        }

        #pragma unroll
        for (int i = 0; i < 4; ++i)
            #pragma unroll
            for (int j = 0; j < 2; ++j)
                acc[i][j] = __builtin_amdgcn_wmma_f32_16x16x32_bf16(
                    false, afr[i], false, bfr[j], (short)0, acc[i][j], false, false);
        __syncthreads();
    }

    // Epilogue. C/D layout: VGPR r -> row = r + 8*(lane/16), col = lane%16.
    #pragma unroll
    for (int i = 0; i < 4; ++i) {
        #pragma unroll
        for (int j = 0; j < 2; ++j) {
            int col = n0 + wc * 32 + j * 16 + l16;
            if (col >= N) continue;
            float bv = bias ? bias[col] : 0.f;
            #pragma unroll
            for (int r = 0; r < 8; ++r) {
                int row = m0 + wr * 64 + i * 16 + half * 8 + r;
                float v = acc[i][j][r] * alpha + bv;
                if (act == 1) v = 0.5f * v * (1.f + erff(v * 0.70710678118f));
                if (resid) {
                    int bidx = row / rowsPerGate;
                    v = resid[(size_t)row * ldc + col] +
                        gate[(size_t)bidx * gateLd + col] * v;
                }
                if (Cf) Cf[(size_t)row * ldc + col] = v;
                else    Cb[(size_t)row * ldc + col] = (bf16_t)v;
            }
        }
    }
}

extern "C" void kernel_launch(void* const* d_in, const int* in_sizes, int n_in,
                              void* d_out, int out_size, void* d_ws, size_t ws_size,
                              hipStream_t stream) {
    (void)in_sizes; (void)n_in; (void)out_size; (void)ws_size;
    const float* x     = (const float*)d_in[0];
    const float* t     = (const float*)d_in[1];
    const float* w_qkv = (const float*)d_in[2];
    const float* b_qkv = (const float*)d_in[3];
    const float* w_m1  = (const float*)d_in[4];
    const float* b_m1  = (const float*)d_in[5];
    const float* w_m2  = (const float*)d_in[6];
    const float* b_m2  = (const float*)d_in[7];
    const float* w_ss1 = (const float*)d_in[8];
    const float* b_ss1 = (const float*)d_in[9];
    const float* w_ss2 = (const float*)d_in[10];
    const float* b_ss2 = (const float*)d_in[11];
    const float* ln1_g = (const float*)d_in[12];
    const float* ln1_b = (const float*)d_in[13];
    const float* ln2_g = (const float*)d_in[14];
    const float* ln2_b = (const float*)d_in[15];
    const float* w_f1  = (const float*)d_in[16];
    const float* b_f1  = (const float*)d_in[17];
    const float* w_f2  = (const float*)d_in[18];
    const float* b_f2  = (const float*)d_in[19];
    float* out = (float*)d_out;

    const int R = B_ * T_;   // 4096 rows

    // ---- workspace carve (all 256-B aligned) ----
    char* wp = (char*)d_ws;
    auto take = [&](size_t bytes) -> void* {
        void* q = (void*)wp;
        wp += (bytes + 255) & ~(size_t)255;
        return q;
    };
    bf16_t* wqkvT = (bf16_t*)take((size_t)H_ * 3 * H_ * 2);   // [3H, H]
    bf16_t* wm1T  = (bf16_t*)take((size_t)H_ * 4 * H_ * 2);   // [4H, H]
    bf16_t* wm2T  = (bf16_t*)take((size_t)4 * H_ * H_ * 2);   // [H, 4H]
    bf16_t* wf1T  = (bf16_t*)take((size_t)H_ * 4 * H_ * 2);   // [4H, H]
    bf16_t* wf2T  = (bf16_t*)take((size_t)4 * H_ * H_ * 2);   // [H, 4H]
    float*  t_hid = (float*) take((size_t)B_ * 6 * H_ * 4);
    float*  t_emb = (float*) take((size_t)B_ * 6 * H_ * 4);
    bf16_t* h_bf  = (bf16_t*)take((size_t)R * H_ * 2);        // h1, then reused for h2
    bf16_t* qkv_bf= (bf16_t*)take((size_t)R * 3 * H_ * 2);    // q,k used in place
    bf16_t* vT_bf = (bf16_t*)take((size_t)R * H_ * 2);        // [BH, HD, T]
    bf16_t* o_bf  = (bf16_t*)take((size_t)R * H_ * 2);
    float*  wei   = (float*) take((size_t)T_ * T_ * 4);       // one head at a time
    bf16_t* p_bf  = (bf16_t*)take((size_t)T_ * T_ * 2);
    bf16_t* big_bf= (bf16_t*)take((size_t)R * 4 * H_ * 2);    // mffn1 out, then ffn1 out
    float*  x1    = (float*) take((size_t)R * H_ * 4);

    auto convT = [&](const float* s, bf16_t* d, int K, int N) {
        size_t n = (size_t)K * N;
        k_f2bf_t<<<(unsigned)((n + 255) / 256), 256, 0, stream>>>(s, d, K, N);
    };
    auto gemm = [&](const bf16_t* A, int lda, const bf16_t* BT, int ldbt,
                    float* Cf, bf16_t* Cb, int ldc, const float* bias,
                    float alpha, int act, const float* resid, const float* gatep,
                    int M, int N, int K) {
        dim3 grid((N + 127) / 128, (M + 127) / 128);
        k_gemm_bf16<<<grid, 256, 0, stream>>>(A, lda, BT, ldbt, Cf, Cb, ldc, bias,
                                              alpha, act, resid, gatep,
                                              6 * H_, T_, M, N, K);
    };

    // 1) weights -> bf16, transposed to [N][K]
    convT(w_qkv, wqkvT, H_, 3 * H_);
    convT(w_m1,  wm1T,  H_, 4 * H_);
    convT(w_m2,  wm2T,  4 * H_, H_);
    convT(w_f1,  wf1T,  H_, 4 * H_);
    convT(w_f2,  wf2T,  4 * H_, H_);

    // 2) adaLN gates: t_emb = silu(t @ w_ss1 + b_ss1) @ w_ss2 + b_ss2  (fp32, tiny)
    k_small_mm<<<(B_ * 6 * H_ + 255) / 256, 256, 0, stream>>>(t, w_ss1, b_ss1, t_hid,
                                                              B_, H_, 6 * H_, 1);
    k_small_mm<<<(B_ * 6 * H_ + 255) / 256, 256, 0, stream>>>(t_hid, w_ss2, b_ss2, t_emb,
                                                              B_, 6 * H_, 6 * H_, 0);

    // 3) h1 = g1 * LN1(x) + be1  -> bf16
    k_ln_mod<<<R, 256, 0, stream>>>(x, ln1_g, ln1_b, t_emb, 0, H_, h_bf);

    // 4) qkv = h1 @ w_qkv + b_qkv  -> bf16  [R, 3H]
    gemm(h_bf, H_, wqkvT, H_, nullptr, qkv_bf, 3 * H_, b_qkv,
         1.f, 0, nullptr, nullptr, R, 3 * H_, H_);

    // 5) build vT (q,k are used in place as strided views of qkv)
    {
        size_t tot = (size_t)R * H_;
        k_repack_v<<<(unsigned)((tot + 255) / 256), 256, 0, stream>>>(qkv_bf, vT_bf);
    }

    // 6) attention, one (b,h) at a time (bounds wei scratch to 16 MB)
    const float scale = 1.0f / sqrtf((float)H_);   // faithful: 1/sqrt(hidden), not head_dim
    for (int bh = 0; bh < B_ * NH_; ++bh) {
        int bb = bh / NH_, hh = bh % NH_;
        const bf16_t* qs = qkv_bf + (size_t)bb * T_ * (3 * H_) + (size_t)hh * HD_;
        const bf16_t* ks = qs + H_;                       // k slice, same strides
        const bf16_t* vTs = vT_bf + (size_t)bh * HD_ * T_;
        bf16_t*       os  = o_bf + (size_t)bb * T_ * H_ + (size_t)hh * HD_;
        // S = scale * q @ k^T : BT = k in natural [t][d] layout.  [T,T] fp32
        gemm(qs, 3 * H_, ks, 3 * H_, wei, nullptr, T_, nullptr, scale, 0,
             nullptr, nullptr, T_, T_, HD_);
        // P = softmax_rows(S) -> bf16
        k_softmax_row<<<T_, 256, 0, stream>>>(wei, p_bf, T_);
        // O = P @ v : BT = vT [HD, T].  -> bf16, strided into [B,T,H]
        gemm(p_bf, T_, vTs, T_, nullptr, os, H_, nullptr, 1.f, 0,
             nullptr, nullptr, T_, HD_, T_);
    }

    // 7) MHA FFN: m1 = gelu(o @ w_m1 + b_m1); x1 = x + a1 * (m1 @ w_m2 + b_m2)
    gemm(o_bf, H_, wm1T, H_, nullptr, big_bf, 4 * H_, b_m1,
         1.f, 1, nullptr, nullptr, R, 4 * H_, H_);
    gemm(big_bf, 4 * H_, wm2T, 4 * H_, x1, nullptr, H_, b_m2,
         1.f, 0, x, t_emb + 2 * H_, R, H_, 4 * H_);

    // 8) h2 = g2 * LN2(x1) + be2 -> bf16 (reuse h buffer)
    k_ln_mod<<<R, 256, 0, stream>>>(x1, ln2_g, ln2_b, t_emb, 3 * H_, 4 * H_, h_bf);

    // 9) DiT FFN: f1 = gelu(h2 @ w_f1 + b_f1); out = x1 + a2 * (f1 @ w_f2 + b_f2)
    gemm(h_bf, H_, wf1T, H_, nullptr, big_bf, 4 * H_, b_f1,
         1.f, 1, nullptr, nullptr, R, 4 * H_, H_);
    gemm(big_bf, 4 * H_, wf2T, 4 * H_, out, nullptr, H_, b_f2,
         1.f, 0, x1, t_emb + 5 * H_, R, H_, 4 * H_);
}
